// P2SAttn_67147518705775
// MI455X (gfx1250) — matrix operand order repaired and checked by reading, not compiled, not run on hardware
//
#include <hip/hip_runtime.h>
#include <hip/hip_bf16.h>
#include <cmath>

// ---------------------------------------------------------------------------
// P2SAttn forward for MI455X (gfx1250, wave32).
// Shapes: B=4, M=1024, N=8192, D=512, G=33 (padded to 64 for WMMA).
//
// Pipeline:
//   1. cast fp32 -> bf16 (A = node_*, B = xyz_*; fpfh padded 33->64 with 0)
//   2. S_f, S_g GEMMs via v_wmma_f32_16x16x32_bf16, fp32 accum, scaled
//   3. Sinkhorn as potentials only:  R = rowLSE(S - C); C = colLSE(S - R)
//      (matrices never rewritten; row potentials cancel in the output)
//   4. per row: z = (S_g - Cg) + (S_f - Cf); iterative top-20 argmax in LDS;
//      softmax-renormalized weights; weighted gather of feats/fpfhs/xyz.
// ---------------------------------------------------------------------------

typedef __attribute__((ext_vector_type(8)))  __bf16 v8bf;
typedef __attribute__((ext_vector_type(16))) __bf16 v16bf;
typedef __attribute__((ext_vector_type(8)))  float  v8f;

#define TOPK 20

// ---------------------------------------------------------------- cast / pad
__global__ __launch_bounds__(256) void pad_cast_kernel(
    const float* __restrict__ in, __hip_bfloat16* __restrict__ out,
    long rows, int Kin, int Kout) {
  long i = (long)blockIdx.x * blockDim.x + threadIdx.x;
  long total = rows * (long)Kout;
  if (i >= total) return;
  long r = i / Kout;
  int  c = (int)(i - r * Kout);
  float v = (c < Kin) ? in[r * (long)Kin + c] : 0.0f;
  out[i] = __float2bfloat16(v);
}

__global__ __launch_bounds__(256) void zero_kernel(float* __restrict__ p, long n) {
  long i = (long)blockIdx.x * blockDim.x + threadIdx.x;
  if (i < n) p[i] = 0.0f;
}

// ------------------------------------------------------------- bf16 WMMA GEMM
// A: [Bb, M, K] row-major (bf16), Bm: [Bb, N, K] row-major (bf16, i.e. B^T),
// S: [Bb, M, N] fp32.  Workgroup = 8 waves = 128x128 tile; wave = 32x64.
__global__ __launch_bounds__(256) void gemm_bf16_wmma(
    const __hip_bfloat16* __restrict__ A, const __hip_bfloat16* __restrict__ Bm,
    float* __restrict__ S, int M, int N, int K, float scale) {
  const int b    = blockIdx.z;
  const int lane = threadIdx.x & 31;
  const int wave = threadIdx.x >> 5;
  const int lm   = lane & 15;   // column / row index inside 16-wide tile
  const int half = lane >> 4;   // which K/M half this lane covers

  const int m0 = blockIdx.y * 128 + (wave & 3) * 32;   // 2 M-tiles of 16
  const int n0 = blockIdx.x * 128 + (wave >> 2) * 64;  // 4 N-tiles of 16

  const __bf16* Ab = (const __bf16*)(A  + (size_t)b * M * K);
  const __bf16* Bb = (const __bf16*)(Bm + (size_t)b * N * K);

  v8f acc[2][4];
#pragma unroll
  for (int mi = 0; mi < 2; ++mi)
#pragma unroll
    for (int ni = 0; ni < 4; ++ni) acc[mi][ni] = {};

  for (int k0 = 0; k0 < K; k0 += 32) {
    v16bf afrag[2], bfrag[4];
    // A 16x32 bf16 fragment: lane half 0 -> K[k0..k0+7],K[k0+16..k0+23];
    //                        lane half 1 -> +8 on both chunks.
#pragma unroll
    for (int mi = 0; mi < 2; ++mi) {
      const v8bf* p = (const v8bf*)(Ab + (size_t)(m0 + mi * 16 + lm) * K + k0 + half * 8);
      afrag[mi] = __builtin_shufflevector(p[0], p[2],
          0, 1, 2, 3, 4, 5, 6, 7, 8, 9, 10, 11, 12, 13, 14, 15);
    }
    // B 32x16 bf16 fragment: lane holds column n0+ni*16+lm, K same as A:
    // [k0 + half*8 .. +7], [k0 + 16 + half*8 .. +7]  (xyz_* rows are K-contiguous: no transpose needed).
#pragma unroll
    for (int ni = 0; ni < 4; ++ni) {
      const v8bf* q = (const v8bf*)(Bb + (size_t)(n0 + ni * 16 + lm) * K + k0 + half * 8);
      bfrag[ni] = __builtin_shufflevector(q[0], q[2],
          0, 1, 2, 3, 4, 5, 6, 7, 8, 9, 10, 11, 12, 13, 14, 15);
    }
#pragma unroll
    for (int mi = 0; mi < 2; ++mi)
#pragma unroll
      for (int ni = 0; ni < 4; ++ni)
        acc[mi][ni] = __builtin_amdgcn_wmma_f32_16x16x32_bf16(
            false, afrag[mi], false, bfrag[ni], (short)0, acc[mi][ni], false, false);
  }

  float* Sb = S + (size_t)b * M * N;
#pragma unroll
  for (int mi = 0; mi < 2; ++mi)
#pragma unroll
    for (int ni = 0; ni < 4; ++ni)
#pragma unroll
      for (int i = 0; i < 8; ++i) {
        const int r = m0 + mi * 16 + half * 8 + i;   // C/D layout: VGPR i -> M
        const int c = n0 + ni * 16 + lm;             // lane%16     -> N
        Sb[(size_t)r * N + c] = acc[mi][ni][i] * scale;
      }
}

// -------------------------------------------------------- Sinkhorn potentials
// R[b,m] = logsumexp_n( S[b,m,n] - C[b,n] )   (one workgroup per row)
__global__ __launch_bounds__(256) void row_lse_kernel(
    const float* __restrict__ S, const float* __restrict__ C,
    float* __restrict__ R, int M, int N) {
  const int b = blockIdx.y, m = blockIdx.x, tid = threadIdx.x;
  const float* row = S + ((size_t)b * M + m) * N;
  const float* Cb  = C + (size_t)b * N;
  float mx = -INFINITY, sm = 0.0f;
  for (int n = tid; n < N; n += 256) {
    float x = row[n] - Cb[n];
    if (x > mx) { sm = sm * __expf(mx - x) + 1.0f; mx = x; }
    else        { sm += __expf(x - mx); }
  }
  __shared__ float smx[256], ssm[256];
  smx[tid] = mx; ssm[tid] = sm;
  __syncthreads();
  for (int off = 128; off > 0; off >>= 1) {
    if (tid < off) {
      float m1 = smx[tid], s1 = ssm[tid];
      float m2 = smx[tid + off], s2 = ssm[tid + off];
      float mm = fmaxf(m1, m2);
      smx[tid] = mm;
      ssm[tid] = s1 * __expf(m1 - mm) + s2 * __expf(m2 - mm);
    }
    __syncthreads();
  }
  if (tid == 0) R[(size_t)b * M + m] = smx[0] + __logf(ssm[0]);
}

// Column LSE split into M-chunks for parallelism (N*B columns alone is only
// 32K threads). partial (max,sum) per chunk, then combine.
__global__ __launch_bounds__(256) void col_lse_partial_kernel(
    const float* __restrict__ S, const float* __restrict__ R,
    float* __restrict__ pmx, float* __restrict__ psm, int M, int N, int MC) {
  const int b = blockIdx.z, chunk = blockIdx.y;
  const int n = blockIdx.x * 256 + threadIdx.x;
  const int mLen = M / MC, mBeg = chunk * mLen;
  const float* Sb = S + (size_t)b * M * N;
  const float* Rb = R + (size_t)b * M;
  float mx = -INFINITY, sm = 0.0f;
  for (int m = mBeg; m < mBeg + mLen; ++m) {
    __builtin_prefetch(Sb + (size_t)(m + 8) * N + n, 0, 0);  // global_prefetch_b8
    float x = Sb[(size_t)m * N + n] - Rb[m];
    if (x > mx) { sm = sm * __expf(mx - x) + 1.0f; mx = x; }
    else        { sm += __expf(x - mx); }
  }
  const size_t o = ((size_t)b * MC + chunk) * N + n;
  pmx[o] = mx; psm[o] = sm;
}

__global__ __launch_bounds__(256) void col_lse_combine_kernel(
    const float* __restrict__ pmx, const float* __restrict__ psm,
    float* __restrict__ C, int N, int MC) {
  const int b = blockIdx.y;
  const int n = blockIdx.x * 256 + threadIdx.x;
  float mx = -INFINITY, sm = 0.0f;
  for (int c = 0; c < MC; ++c) {
    const size_t o = ((size_t)b * MC + c) * N + n;
    float m2 = pmx[o], s2 = psm[o];
    float mm = fmaxf(mx, m2);
    sm = sm * __expf(mx - mm) + s2 * __expf(m2 - mm);
    mx = mm;
  }
  C[(size_t)b * N + n] = mx + __logf(sm);
}

// ----------------------------------------------------- top-k gather finalize
__global__ __launch_bounds__(256) void finalize_kernel(
    const float* __restrict__ Sg, const float* __restrict__ Sf,
    const float* __restrict__ Cg, const float* __restrict__ Cf,
    const float* __restrict__ xyz_feats, const float* __restrict__ xyz_fpfhs,
    const float* __restrict__ xyz,
    const float* __restrict__ node_feats, const float* __restrict__ node_fpfhs,
    const float* __restrict__ node_xyz,
    float* __restrict__ out_feats, float* __restrict__ out_fpfhs,
    float* __restrict__ out_xyz, int M, int N, int D, int G) {
  const int b = blockIdx.y, m = blockIdx.x, tid = threadIdx.x;
  const size_t rowOff = ((size_t)b * M + m) * N;

  __shared__ float z[8192];       // 32 KB of the WGP's 320 KB LDS
  __shared__ float sv[256];
  __shared__ int   si[256];
  __shared__ int   topIdx[TOPK];
  __shared__ float topVal[TOPK];
  __shared__ float wk[TOPK];
  __shared__ float wsum;

  // z[n] ∝ log attn[b,m,n] up to a per-row constant (row potentials cancel).
  for (int n = tid; n < N; n += 256)
    z[n] = (Sg[rowOff + n] - Cg[(size_t)b * N + n]) +
           (Sf[rowOff + n] - Cf[(size_t)b * N + n]);
  __syncthreads();

  // Iterative argmax (lowest-index tie-break, matching lax.top_k).
  for (int k = 0; k < TOPK; ++k) {
    float best = -INFINITY; int bidx = 0x7FFFFFFF;
    for (int n = tid; n < N; n += 256) {
      float v = z[n];
      if (v > best) { best = v; bidx = n; }   // ascending scan -> first max kept
    }
    sv[tid] = best; si[tid] = bidx;
    __syncthreads();
    for (int off = 128; off > 0; off >>= 1) {
      if (tid < off) {
        float v2 = sv[tid + off]; int i2 = si[tid + off];
        if (v2 > sv[tid] || (v2 == sv[tid] && i2 < si[tid])) {
          sv[tid] = v2; si[tid] = i2;
        }
      }
      __syncthreads();
    }
    if (tid == 0) { topIdx[k] = si[0]; topVal[k] = sv[0]; z[si[0]] = -INFINITY; }
    __syncthreads();
  }

  // w_k = exp(z_k) / sum_j exp(z_j): row constants cancel -> softmax over top-k.
  if (tid < TOPK) wk[tid] = __expf(topVal[tid] - topVal[0]);
  __syncthreads();
  if (tid == 0) {
    float s = 0.0f;
    for (int k = 0; k < TOPK; ++k) s += wk[k];
    wsum = s;
  }
  __syncthreads();
  const float inv = 1.0f / wsum;

  const int total = D + G + 3;  // 548 output channels per prototype
  for (int c = tid; c < total; c += 256) {
    const float* src; const float* nodev; float* dst; int stride, col;
    if (c < D) {
      src = xyz_feats; stride = D; col = c;
      nodev = node_feats + ((size_t)b * M + m) * D + col;
      dst   = out_feats  + ((size_t)b * M + m) * D + col;
    } else if (c < D + G) {
      src = xyz_fpfhs; stride = G; col = c - D;
      nodev = node_fpfhs + ((size_t)b * M + m) * G + col;
      dst   = out_fpfhs  + ((size_t)b * M + m) * G + col;
    } else {
      src = xyz; stride = 3; col = c - D - G;
      nodev = node_xyz + ((size_t)b * M + m) * 3 + col;
      dst   = out_xyz  + ((size_t)b * M + m) * 3 + col;
    }
    float acc = 0.0f;
#pragma unroll
    for (int k = 0; k < TOPK; ++k)
      acc += wk[k] * inv * src[((size_t)b * N + topIdx[k]) * stride + col];
    *dst = (acc + *nodev) * 0.5f;
  }
}

// ----------------------------------------------------------------- launcher
extern "C" void kernel_launch(void* const* d_in, const int* in_sizes, int n_in,
                              void* d_out, int out_size, void* d_ws, size_t ws_size,
                              hipStream_t stream) {
  const float* node_xyz   = (const float*)d_in[0];  // [4,1024,3]
  const float* xyz        = (const float*)d_in[1];  // [4,8192,3]
  const float* node_feats = (const float*)d_in[2];  // [4,1024,512]
  const float* xyz_feats  = (const float*)d_in[3];  // [4,8192,512]
  const float* node_fpfhs = (const float*)d_in[4];  // [4,1024,33]
  const float* xyz_fpfhs  = (const float*)d_in[5];  // [4,8192,33]

  const int B = 4, M = 1024, N = 8192, D = 512, G = 33, Gp = 64, MC = 8;
  const int NITERS = 5;

  // workspace carve-out (~314 MB total)
  char* ws = (char*)d_ws;
  size_t off = 0;
  auto carve = [&](size_t bytes) -> void* {
    void* p = ws + off;
    off = (off + bytes + 255) & ~(size_t)255;
    return p;
  };
  float*           Sg  = (float*)carve((size_t)B * M * N * 4);          // 134 MB
  float*           Sf  = (float*)carve((size_t)B * M * N * 4);          // 134 MB
  __hip_bfloat16*  Afc = (__hip_bfloat16*)carve((size_t)B * M * D  * 2);
  __hip_bfloat16*  Bfc = (__hip_bfloat16*)carve((size_t)B * N * D  * 2);
  __hip_bfloat16*  Agc = (__hip_bfloat16*)carve((size_t)B * M * Gp * 2);
  __hip_bfloat16*  Bgc = (__hip_bfloat16*)carve((size_t)B * N * Gp * 2);
  float*           Cg  = (float*)carve((size_t)B * N * 4);
  float*           Cf  = (float*)carve((size_t)B * N * 4);
  float*           Rg  = (float*)carve((size_t)B * M * 4);
  float*           Rf  = (float*)carve((size_t)B * M * 4);
  float*           pmx = (float*)carve((size_t)B * MC * N * 4);
  float*           psm = (float*)carve((size_t)B * MC * N * 4);

  // 1) casts (pad fpfh K 33 -> 64 with zeros so K % 32 == 0)
  {
    long t;
    t = (long)B * M * D;
    pad_cast_kernel<<<(t + 255) / 256, 256, 0, stream>>>(node_feats, Afc, (long)B * M, D, D);
    t = (long)B * N * D;
    pad_cast_kernel<<<(t + 255) / 256, 256, 0, stream>>>(xyz_feats, Bfc, (long)B * N, D, D);
    t = (long)B * M * Gp;
    pad_cast_kernel<<<(t + 255) / 256, 256, 0, stream>>>(node_fpfhs, Agc, (long)B * M, G, Gp);
    t = (long)B * N * Gp;
    pad_cast_kernel<<<(t + 255) / 256, 256, 0, stream>>>(xyz_fpfhs, Bgc, (long)B * N, Gp == G ? G : G, Gp);
  }

  // 2) WMMA GEMMs (scaled scores written once; never rewritten)
  const float sf = 1.0f / sqrtf((float)D);
  const float sg = 1.0f / sqrtf((float)G);
  gemm_bf16_wmma<<<dim3(N / 128, M / 128, B), 256, 0, stream>>>(Afc, Bfc, Sf, M, N, D,  sf);
  gemm_bf16_wmma<<<dim3(N / 128, M / 128, B), 256, 0, stream>>>(Agc, Bgc, Sg, M, N, Gp, sg);

  // 3) Sinkhorn as row/column potentials
  zero_kernel<<<((long)B * N + 255) / 256, 256, 0, stream>>>(Cg, (long)B * N);
  zero_kernel<<<((long)B * N + 255) / 256, 256, 0, stream>>>(Cf, (long)B * N);
  for (int it = 0; it < NITERS; ++it) {
    row_lse_kernel<<<dim3(M, B), 256, 0, stream>>>(Sg, Cg, Rg, M, N);
    col_lse_partial_kernel<<<dim3(N / 256, MC, B), 256, 0, stream>>>(Sg, Rg, pmx, psm, M, N, MC);
    col_lse_combine_kernel<<<dim3(N / 256, B), 256, 0, stream>>>(pmx, psm, Cg, N, MC);

    row_lse_kernel<<<dim3(M, B), 256, 0, stream>>>(Sf, Cf, Rf, M, N);
    col_lse_partial_kernel<<<dim3(N / 256, MC, B), 256, 0, stream>>>(Sf, Rf, pmx, psm, M, N, MC);
    col_lse_combine_kernel<<<dim3(N / 256, B), 256, 0, stream>>>(pmx, psm, Cf, N, MC);
  }

  // 4) top-20 + weighted gather; outputs concatenated in return order
  float* out_feats = (float*)d_out;                       // [4,1024,512]
  float* out_fpfhs = out_feats + (size_t)B * M * D;       // [4,1024,33]
  float* out_xyz   = out_fpfhs + (size_t)B * M * G;       // [4,1024,3]
  finalize_kernel<<<dim3(M, B), 256, 0, stream>>>(
      Sg, Sf, Cg, Cf, xyz_feats, xyz_fpfhs, xyz,
      node_feats, node_fpfhs, node_xyz,
      out_feats, out_fpfhs, out_xyz, M, N, D, G);
}
